// LSTM_43877385896198
// MI455X (gfx1250) — matrix operand order repaired and checked by reading, not compiled, not run on hardware
//
#include <hip/hip_runtime.h>

// ---------------------------------------------------------------------------
// Lattice LSTM (T=8192 nodes, W=4 edges, I=H=512) for MI455X / gfx1250.
//
// Phase 0: convert x, w_ih, w_hh to f16 in workspace (HBM streams).
// Phase 1: gates_x[32768,2048] = x@w_ih^T + (b_ih+b_hh): f16 WMMA, f32 acc,
//          software-pipelined (double-buffered A/B fragments) so WMMAs
//          overlap the L2-latency of fragment loads.
// Phase 2: 16 persistent WGs; each pins 128 rows of w_hh (128KB) in LDS
//          (320KB/WGP), owns h-slice [32w,32w+32), computes its 4 gate rows
//          locally, 1 global barrier per step, h ping-pongs through L2.
// ---------------------------------------------------------------------------

typedef _Float16 f16;
typedef __attribute__((ext_vector_type(16))) _Float16 v16h;
typedef __attribute__((ext_vector_type(8)))  _Float16 v8h;
typedef __attribute__((ext_vector_type(4)))  _Float16 v4h;
typedef __attribute__((ext_vector_type(4)))  float    v4f;
typedef __attribute__((ext_vector_type(8)))  float    v8f;

#define T_STEPS 8192
#define W_E     4
#define I_DIM   512
#define H_DIM   512
#define M_TOT   (T_STEPS * W_E)   // 32768 rows of gates_x
#define N_TOT   (4 * H_DIM)       // 2048 gate columns
#define K_TOT   I_DIM             // 512

// ---- workspace layout (bytes) ----
#define BAR_OFF  0                         // arrive counter (memset each launch)
#define HBUF_OFF 1024                      // float hbuf[2][512] (memset each launch)
#define ZERO_BYTES 8192
#define XF16_OFF  8192                                    // f16 x  [32768][512]
#define WIH_OFF  (XF16_OFF + (size_t)M_TOT * K_TOT * 2)   // f16 w_ih [2048][512]
#define WHH_OFF  (WIH_OFF  + (size_t)N_TOT * K_TOT * 2)   // f16 w_hh [2048][512]
#define GX_OFF   (WHH_OFF  + (size_t)N_TOT * K_TOT * 2)   // f16 gates_x [32768][2048]

__global__ void cvt_f32_to_f16(const float* __restrict__ src,
                               f16* __restrict__ dst, int n4) {
    // n4 = element count / 4 (all our sizes are multiples of 4)
    int i = blockIdx.x * blockDim.x + threadIdx.x;
    int stride = gridDim.x * blockDim.x;
    for (; i < n4; i += stride) {
        v4f v = *(const v4f*)(src + (size_t)i * 4);
        v4h h;
#pragma unroll
        for (int q = 0; q < 4; ++q) h[q] = (f16)v[q];
        *(v4h*)(dst + (size_t)i * 4) = h;
    }
}

// A fragment, 16x32 f16: lane half lh holds 16 contiguous k:
// k = kk + lh*16 + 0..15 (matches the B fragment order).
__device__ __forceinline__ v16h load_a_frag(const f16* __restrict__ arow,
                                            int kk, int lh) {
    v8h alo = *(const v8h*)(arow + kk + lh * 16);
    v8h ahi = *(const v8h*)(arow + kk + lh * 16 + 8);
    v16h a;
#pragma unroll
    for (int i = 0; i < 8; ++i) { a[i] = alo[i]; a[8 + i] = ahi[i]; }
    return a;
}

// One wave computes a 16x128 strip of gates_x: 8 WMMA tiles along N sharing
// one A fragment; K looped in steps of 32. Double-buffered fragments so the
// 8 WMMAs of iteration k hide the 18 loads of iteration k+1.
__launch_bounds__(256)
__global__ void gemm_gatesx(const f16* __restrict__ A,    // [M_TOT][K_TOT]
                            const f16* __restrict__ Wih,  // [N_TOT][K_TOT]
                            const float* __restrict__ bih,
                            const float* __restrict__ bhh,
                            f16* __restrict__ GX)         // [M_TOT][N_TOT]
{
    const int lane = threadIdx.x & 31;
    const int wave = threadIdx.x >> 5;
    const int wid  = blockIdx.x * 8 + wave;   // 0..32767
    const int mtile  = wid >> 4;              // 0..2047
    const int nstrip = wid & 15;              // 0..15
    const int m0 = mtile * 16;
    const int n0 = nstrip * 128;

    const int lh = lane >> 4;                 // lane half (0/1)
    const int lm = lane & 15;

    v8f acc[8] = {};

    const f16* arow  = A + (size_t)(m0 + lm) * K_TOT;
    // B frag 32x16: B[k][n] = w_ih[n][k]; lane holds n = lm, 16 contiguous k
    // starting at kk + lh*16 -> one 32B contiguous run per fragment.
    const f16* bbase = Wih + (size_t)(n0 + lm) * K_TOT + lh * 16;

    v16h a_cur = load_a_frag(arow, 0, lh);
    v16h b_cur[8];
#pragma unroll
    for (int u = 0; u < 8; ++u)
        b_cur[u] = *(const v16h*)(bbase + (size_t)u * 16 * K_TOT);

    for (int kk = 0; kk < K_TOT - 32; kk += 32) {
        // prefetch next iteration's fragments into the second buffer
        v16h a_nxt = load_a_frag(arow, kk + 32, lh);
        v16h b_nxt[8];
#pragma unroll
        for (int u = 0; u < 8; ++u)
            b_nxt[u] = *(const v16h*)(bbase + (kk + 32)
                                      + (size_t)u * 16 * K_TOT);
        // consume current fragments (covers the in-flight loads)
#pragma unroll
        for (int u = 0; u < 8; ++u)
            acc[u] = __builtin_amdgcn_wmma_f32_16x16x32_f16(
                false, a_cur, false, b_cur[u], (short)0, acc[u], false, false);
        a_cur = a_nxt;
#pragma unroll
        for (int u = 0; u < 8; ++u) b_cur[u] = b_nxt[u];
    }
    // final K block
#pragma unroll
    for (int u = 0; u < 8; ++u)
        acc[u] = __builtin_amdgcn_wmma_f32_16x16x32_f16(
            false, a_cur, false, b_cur[u], (short)0, acc[u], false, false);

    // Epilogue: D element r -> m = m0 + lh*8 + r, n = n0 + u*16 + lm.
#pragma unroll
    for (int u = 0; u < 8; ++u) {
        const int n = n0 + u * 16 + lm;
        const float bias = bih[n] + bhh[n];
#pragma unroll
        for (int r = 0; r < 8; ++r) {
            const int m = m0 + lh * 8 + r;
            GX[(size_t)m * N_TOT + n] = (f16)(acc[u][r] + bias);
        }
    }
}

#if __has_builtin(__builtin_amdgcn_global_load_async_to_lds_b128) && \
    __has_builtin(__builtin_amdgcn_s_wait_asynccnt)
#define HAVE_ASYNC_LDS 1
// Param types per hipcc diagnostic: pointer to int __attribute__((vector_size(16)))
// in AS1 (global, prints as __device__) / AS3 (LDS) respectively.
typedef int v4i __attribute__((vector_size(4 * sizeof(int))));
typedef __attribute__((address_space(1))) v4i* gv4i_p;
typedef __attribute__((address_space(3))) v4i* lv4i_p;
#else
#define HAVE_ASYNC_LDS 0
#endif

// Persistent recurrence: grid = 16 WGs x 256 threads. WG w owns h indices
// [32w, 32w+32) and gate rows {g*512 + 32w + jj}. Weights stay in LDS for
// all 8192 steps; c-state stays in LDS; only h (2KB) crosses WGs via L2.
__launch_bounds__(256)
__global__ void lstm_recur(const f16* __restrict__ GX,    // [M_TOT][2048]
                           const f16* __restrict__ Whh,   // [2048][512]
                           float* __restrict__ out,       // [M_TOT][512]
                           float* __restrict__ hbuf,      // [2][512], zeroed
                           unsigned int* __restrict__ cnt)// zeroed
{
    extern __shared__ unsigned char smem[];
    f16*   whh_s   = (f16*)smem;                         // [128][512] = 128KB
    float* h_s     = (float*)(smem + 128 * 512 * 2);     // [512]
    float* gates_s = h_s + 512;                          // [128]
    float* he_s    = gates_s + 128;                      // [4][32]
    float* ce_s    = he_s + 128;                         // [4][32]
    float* c_s     = ce_s + 128;                         // [32]

    const int tid = threadIdx.x;
    const int wg  = blockIdx.x;       // 0..15
    const int j0  = wg * 32;

    // Stage this WG's 128 w_hh rows: LDS row r = g*32+jj <- global g*512+j0+jj
#if HAVE_ASYNC_LDS
    {
        f16* whh_nc = const_cast<f16*>(Whh);
        for (int idx = tid; idx < (128 * 512) / 8; idx += 256) {
            const int flat = idx * 8;             // f16 elements (16B chunks)
            const int r = flat >> 9;
            const int k = flat & 511;
            const int grow = (r >> 5) * 512 + j0 + (r & 31);
            __builtin_amdgcn_global_load_async_to_lds_b128(
                (gv4i_p)(whh_nc + (size_t)grow * 512 + k),
                (lv4i_p)(whh_s + (size_t)r * 512 + k),
                /*offset=*/0, /*cpol=*/0);
        }
        __builtin_amdgcn_s_wait_asynccnt(0);
    }
#else
    for (int idx = tid; idx < (128 * 512) / 4; idx += 256) {
        const int flat = idx * 4;
        const int r = flat >> 9;
        const int k = flat & 511;
        const int grow = (r >> 5) * 512 + j0 + (r & 31);
        *(v4h*)(whh_s + (size_t)r * 512 + k) =
            *(const v4h*)(Whh + (size_t)grow * 512 + k);
    }
#endif
    if (tid < 32) c_s[tid] = 0.0f;
    __syncthreads();

    const int rlocal = tid >> 1;      // gate row 0..127 (g*32+jj)
    const int khalf  = tid & 1;       // k range half
    unsigned int epoch = 0;

    for (int t = 0; t < T_STEPS; ++t) {
        // h for this step (t=0 reads memset zeros)
        const float* hin = hbuf + (size_t)(t & 1) * 512;
        for (int i = tid; i < 512; i += 256) h_s[i] = hin[i];

        // warm L2/L0 with next step's gates_x (16 x 64B regions per WG)
        if (t + 1 < T_STEPS && tid < 16) {
            const int pe = tid >> 2;   // edge
            const int pg = tid & 3;    // gate
            __builtin_prefetch(GX + (size_t)((t + 1) * 4 + pe) * 2048
                               + pg * 512 + j0, 0, 1);
        }
        __syncthreads();

        // GEMV: 2 threads per gate row, 256 MACs each, pair-reduce in wave.
        float acc = 0.0f;
        const f16*  wr = whh_s + (size_t)rlocal * 512 + khalf * 256;
        const float* hp = h_s + khalf * 256;
#pragma unroll 4
        for (int k = 0; k < 256; k += 8) {
            v8h wv = *(const v8h*)(wr + k);
#pragma unroll
            for (int q = 0; q < 8; ++q) acc += (float)wv[q] * hp[k + q];
        }
        acc += __shfl_xor(acc, 1);
        if (khalf == 0) gates_s[rlocal] = acc;
        __syncthreads();

        // Per-edge gate math for our 32 h indices (4 edges x 32 lanes).
        if (tid < 128) {
            const int e  = tid >> 5;
            const int jj = tid & 31;
            const f16* gxr = GX + (size_t)(t * 4 + e) * 2048 + j0 + jj;
            const float ip = gates_s[jj]      + (float)gxr[0];
            const float fp = gates_s[32 + jj] + (float)gxr[512];
            const float gp = gates_s[64 + jj] + (float)gxr[1024];
            const float op = gates_s[96 + jj] + (float)gxr[1536];
            const float ig = 1.0f / (1.0f + __expf(-ip));
            const float fg = 1.0f / (1.0f + __expf(-fp));
            const float gg = tanhf(gp);
            const float og = 1.0f / (1.0f + __expf(-op));
            const float ce = fg * c_s[jj] + ig * gg;
            const float he = og * tanhf(ce);
            out[(size_t)(t * 4 + e) * 512 + j0 + jj] = he;
            he_s[tid] = he;
            ce_s[tid] = ce;
        }
        __syncthreads();

        // combine_edges('mean'); publish next h slice.
        if (tid < 32) {
            const float hm = 0.25f * (he_s[tid] + he_s[32 + tid] +
                                      he_s[64 + tid] + he_s[96 + tid]);
            const float cm = 0.25f * (ce_s[tid] + ce_s[32 + tid] +
                                      ce_s[64 + tid] + ce_s[96 + tid]);
            c_s[tid] = cm;
            hbuf[(size_t)((t + 1) & 1) * 512 + j0 + tid] = hm;
        }
        __threadfence();
        __syncthreads();

        // One device barrier per step: monotonic arrive counter.
        ++epoch;
        if (tid == 0) {
            __hip_atomic_fetch_add(cnt, 1u, __ATOMIC_RELEASE,
                                   __HIP_MEMORY_SCOPE_AGENT);
            const unsigned int target = 16u * epoch;
            while (__hip_atomic_load(cnt, __ATOMIC_ACQUIRE,
                                     __HIP_MEMORY_SCOPE_AGENT) < target) {
                __builtin_amdgcn_s_sleep(1);
            }
        }
        __syncthreads();
    }
}

extern "C" void kernel_launch(void* const* d_in, const int* in_sizes, int n_in,
                              void* d_out, int out_size, void* d_ws, size_t ws_size,
                              hipStream_t stream) {
    const float* x    = (const float*)d_in[0];  // [8192][4][512]
    const float* w_ih = (const float*)d_in[1];  // [2048][512]
    const float* w_hh = (const float*)d_in[2];  // [2048][512]
    const float* b_ih = (const float*)d_in[3];  // [2048]
    const float* b_hh = (const float*)d_in[4];  // [2048]
    float* out = (float*)d_out;                 // [32768][512]

    unsigned char* ws = (unsigned char*)d_ws;
    unsigned int* cnt = (unsigned int*)(ws + BAR_OFF);
    float* hbuf       = (float*)(ws + HBUF_OFF);
    f16* x16   = (f16*)(ws + XF16_OFF);
    f16* wih16 = (f16*)(ws + WIH_OFF);
    f16* whh16 = (f16*)(ws + WHH_OFF);
    f16* gx16  = (f16*)(ws + GX_OFF);

    // Reset barrier counter + h ping-pong (graph-capture safe).
    (void)hipMemsetAsync(ws, 0, ZERO_BYTES, stream);

    cvt_f32_to_f16<<<2048, 256, 0, stream>>>(x,    x16,   (M_TOT * K_TOT) / 4);
    cvt_f32_to_f16<<<512,  256, 0, stream>>>(w_ih, wih16, (N_TOT * K_TOT) / 4);
    cvt_f32_to_f16<<<512,  256, 0, stream>>>(w_hh, whh16, (N_TOT * K_TOT) / 4);

    // 2048 m-tiles x 16 n-strips = 32768 waves = 4096 blocks of 8 waves.
    gemm_gatesx<<<4096, 256, 0, stream>>>(x16, wih16, b_ih, b_hh, gx16);

    const size_t smem = (size_t)128 * 512 * 2          // w_hh slice
                      + (512 + 128 + 128 + 128 + 32) * sizeof(float);
    lstm_recur<<<16, 256, smem, stream>>>(gx16, whh16, out, hbuf, cnt);
}
